// SelfAttention_88708254532129
// MI455X (gfx1250) — matrix operand order repaired and checked by reading, not compiled, not run on hardware
//
#include <hip/hip_runtime.h>
#include <math.h>

// ---------------------------------------------------------------------------
// MI455X (gfx1250) attention block.
// Precision: all GEMM-shaped math through v_wmma_f32_16x16x32_f16 (f16 in,
// f32 accumulate); softmax / RMSNorm statistics in f32.
// Roofline: ~155 GFLOP vs ~130MB unique bytes -> matrix-core bound.
// Data movement strategy:
//   * GEMMs: direct per-lane contiguous b128 fragment loads (L2-resident
//     weight/activation streams; compiler clauses + staggered loadcnt waits).
//   * Flash attention: K/V tiles staged to LDS once per block with
//     global_load_async_to_lds_b128 (ASYNCcnt, double-buffered) -- removes
//     the 8x per-wave redundancy -- with bank-conflict-free padded strides.
// ---------------------------------------------------------------------------

typedef __attribute__((ext_vector_type(16))) _Float16 v16h;
typedef __attribute__((ext_vector_type(8)))  _Float16 v8h;
typedef __attribute__((ext_vector_type(4)))  _Float16 v4h;
typedef __attribute__((ext_vector_type(8)))  float    v8f;

static constexpr int kB   = 2;
static constexpr int kL   = 2048;
static constexpr int kC   = 2048;
static constexpr int kNH  = 16;
static constexpr int kNKV = 4;
static constexpr int kHD  = 128;
static constexpr float kScale = 0.08838834764831845f;  // 1/sqrt(HD)

#define WMMA_F16(a, b, c) \
  __builtin_amdgcn_wmma_f32_16x16x32_f16(false, (a), false, (b), (short)0, (c), false, false)

// ---- CDNA5 async global->LDS path (guarded; fallback = sync copy) ---------
#if defined(__has_builtin)
# if __has_builtin(__builtin_amdgcn_global_load_async_to_lds_b128)
#  define USE_ASYNC_LDS 1
# endif
# if __has_builtin(__builtin_amdgcn_s_wait_asynccnt)
#  define HAVE_WAIT_ASYNC_BUILTIN 1
# endif
#endif

// Builtin signature (from the round-2 diagnostic) uses GCC-vector int4
// pointees with target address spaces: (v4i AS1*, v4i AS3*, imm, imm).
typedef int v4i_gcc __attribute__((vector_size(16)));
typedef v4i_gcc __attribute__((address_space(1)))* gptr_b128;
typedef v4i_gcc __attribute__((address_space(3)))* lptr_b128;

__device__ __forceinline__ void cp16_to_lds(const _Float16* g, _Float16* l) {
#ifdef USE_ASYNC_LDS
  __builtin_amdgcn_global_load_async_to_lds_b128((gptr_b128)g, (lptr_b128)l, 0, 0);
#else
  *(v8h*)l = *(const v8h*)g;   // fallback: global load + ds_store
#endif
}

template <int N>
__device__ __forceinline__ void wait_async_le() {
#ifdef USE_ASYNC_LDS
# ifdef HAVE_WAIT_ASYNC_BUILTIN
  __builtin_amdgcn_s_wait_asynccnt((unsigned short)N);
# else
  asm volatile("s_wait_asynccnt %0" ::"i"(N) : "memory");
# endif
#endif
}

// A-matrix 16x32 f16 fragment (ISA 7.12.2): lanes 0-15 hold row=lane,
// K=0..7 (v0-3) and K=16..23 (v4-7); lanes 16-31 hold K=8..15 / 24..31.
__device__ __forceinline__ v16h load_frag_a(const _Float16* __restrict__ p, int ld) {
  const int lane = threadIdx.x & 31;
  const _Float16* q = p + (size_t)(lane & 15) * ld + ((lane >> 4) << 3);
  v8h lo = *(const v8h*)q;          // K = koff .. koff+7
  v8h hi = *(const v8h*)(q + 16);   // K = koff+16 .. koff+23
  v16h f;
#pragma unroll
  for (int i = 0; i < 8; ++i) { f[i] = lo[i]; f[i + 8] = hi[i]; }
  return f;
}

// B-matrix 32x16 f16 fragment: lanes 0-15 hold col n=lane, K=0..7 / 16..23;
// lanes 16-31 hold K=8..15 / 24..31. Source is row-major [n][k] (K contiguous).
__device__ __forceinline__ v16h load_frag_b(const _Float16* __restrict__ p, int ld) {
  const int lane = threadIdx.x & 31;
  const _Float16* q = p + (size_t)(lane & 15) * ld + ((lane >> 4) << 3);
  v8h lo = *(const v8h*)q;          // K = koff .. koff+7
  v8h hi = *(const v8h*)(q + 16);   // K = koff+16 .. koff+23
  v16h f;
#pragma unroll
  for (int i = 0; i < 8; ++i) { f[i] = lo[i]; f[i + 8] = hi[i]; }
  return f;
}

// ---------------------------------------------------------------------------
// f32 -> f16 convert, 4 elements / thread
// ---------------------------------------------------------------------------
__global__ __launch_bounds__(256)
void cvt_f32_f16_x4(const float* __restrict__ in, _Float16* __restrict__ out, int n4) {
  int i = blockIdx.x * 256 + threadIdx.x;
  if (i >= n4) return;
  float4 v = ((const float4*)in)[i];
  v4h o;
  o[0] = (_Float16)v.x; o[1] = (_Float16)v.y;
  o[2] = (_Float16)v.z; o[3] = (_Float16)v.w;
  ((v4h*)out)[i] = o;
}

// ---------------------------------------------------------------------------
// C[M,N] (f32) = A[M,K] (f16, row major) * B[N,K]^T (f16, row major)
// 128x128 block tile, 8 waves, wave tile 32x64 (2x4 WMMA frags), K step 32.
// ---------------------------------------------------------------------------
__global__ __launch_bounds__(256)
void gemm_nt_wmma(const _Float16* __restrict__ A, const _Float16* __restrict__ Bw,
                  float* __restrict__ C, int M, int N, int K) {
  const int lane = threadIdx.x & 31;
  const int w    = threadIdx.x >> 5;       // 0..7
  const int wm   = w >> 1;                 // 0..3
  const int wn   = w & 1;                  // 0..1
  const int m0 = blockIdx.y * 128 + wm * 32;
  const int n0 = blockIdx.x * 128 + wn * 64;

  v8f acc[2][4] = {};

  const _Float16* a0 = A + (size_t)m0 * K;
  const _Float16* a1 = A + (size_t)(m0 + 16) * K;

  for (int kk = 0; kk < K; kk += 32) {
    if (kk + 32 < K) {  // global_prefetch_b8 of next K tile
      __builtin_prefetch(a0 + (size_t)(lane & 15) * K + kk + 32, 0, 1);
      __builtin_prefetch(Bw + (size_t)(n0 + (lane & 15)) * K + kk + 32, 0, 1);
    }
    v16h af[2], bf[4];
    af[0] = load_frag_a(a0 + kk, K);
    af[1] = load_frag_a(a1 + kk, K);
#pragma unroll
    for (int j = 0; j < 4; ++j)
      bf[j] = load_frag_b(Bw + (size_t)(n0 + 16 * j) * K + kk, K);
#pragma unroll
    for (int i = 0; i < 2; ++i)
#pragma unroll
      for (int j = 0; j < 4; ++j)
        acc[i][j] = WMMA_F16(af[i], bf[j], acc[i][j]);
  }

  // C/D layout: lane half g holds rows 8g..8g+7 in v0..v7, col = n&15
  const int g  = lane >> 4;
  const int nn = lane & 15;
#pragma unroll
  for (int i = 0; i < 2; ++i)
#pragma unroll
    for (int j = 0; j < 4; ++j) {
      float* cp = C + (size_t)(m0 + 16 * i + g * 8) * N + (n0 + 16 * j + nn);
#pragma unroll
      for (int r = 0; r < 8; ++r) cp[(size_t)r * N] = acc[i][j][r];
    }
}

// ---------------------------------------------------------------------------
// RMSNorm (per 128-d head row) + half-split RoPE; f32 in -> f16 out [b,h,l,d]
// ---------------------------------------------------------------------------
__global__ __launch_bounds__(256)
void rmsnorm_rope_kernel(const float* __restrict__ in,        // [B, L, H*128]
                         const float* __restrict__ nw,        // [128]
                         const float* __restrict__ freqs,     // [L, 64]
                         _Float16* __restrict__ out,          // [B*H, L, 128]
                         int H, float outscale) {
  const int lane = threadIdx.x & 31;
  const int w    = threadIdx.x >> 5;
  const size_t row = (size_t)blockIdx.x * 8 + w;   // over B*L*H
  const int    h   = (int)(row % H);
  const size_t bl  = row / H;                      // b*L + l
  const int    l   = (int)(bl % kL);
  const size_t b   = bl / kL;

  const int d0 = lane * 4;
  float4 xv = *(const float4*)(in + row * kHD + d0);
  float x[4] = {xv.x, xv.y, xv.z, xv.w};

  float ss = x[0]*x[0] + x[1]*x[1] + x[2]*x[2] + x[3]*x[3];
#pragma unroll
  for (int d = 1; d < 32; d <<= 1) ss += __shfl_xor(ss, d, 32);
  const float r = rsqrtf(ss * (1.0f / 128.0f) + 1e-6f);

  float nv[4], ov[4];
#pragma unroll
  for (int e = 0; e < 4; ++e) nv[e] = x[e] * r * nw[d0 + e];
#pragma unroll
  for (int e = 0; e < 4; ++e) ov[e] = __shfl_xor(nv[e], 16, 32);  // rotary partner

  const int j0 = d0 & 63;
  v4h pack;
#pragma unroll
  for (int e = 0; e < 4; ++e) {
    float fr = freqs[(size_t)l * 64 + j0 + e];
    float cs = __cosf(fr), sn = __sinf(fr);
    float o = (lane < 16) ? (nv[e] * cs - ov[e] * sn)    // t1*cos - t2*sin
                          : (nv[e] * cs + ov[e] * sn);   // t2*cos + t1*sin
    pack[e] = (_Float16)(o * outscale);
  }
  _Float16* dst = out + (((b * H + h) * (size_t)kL + l) * kHD) + d0;
  *(v4h*)dst = pack;
}

// ---------------------------------------------------------------------------
// V: f32 [B, L, NKV*128] -> f16 transposed [B*NKV, 128, L]
// ---------------------------------------------------------------------------
__global__ __launch_bounds__(256)
void v_transpose_kernel(const float* __restrict__ vf, _Float16* __restrict__ vth) {
  size_t idx = (size_t)blockIdx.x * 256 + threadIdx.x;   // over B*L*NKV*HD
  const int d = (int)(idx % kHD);
  size_t t = idx / kHD;
  const int h = (int)(t % kNKV); t /= kNKV;
  const int l = (int)(t % kL);
  const size_t b = t / kL;
  vth[(((b * kNKV + h) * (size_t)kHD + d) * kL) + l] = (_Float16)vf[idx];
}

// ---------------------------------------------------------------------------
// Flash attention (GQA 4:1). Block = 8 waves = 128 query rows of one (b,h).
// K/V 32-key tiles staged to LDS once per block (async, double-buffered);
// padded strides -> conflict-free ds_load_b128 fragments for all 8 waves.
// ---------------------------------------------------------------------------
__global__ __launch_bounds__(256)
void flash_attn_kernel(const _Float16* __restrict__ qh,   // [B*NH, L, 128] (pre-scaled)
                       const _Float16* __restrict__ kh,   // [B*NKV, L, 128]
                       const _Float16* __restrict__ vth,  // [B*NKV, 128, L]
                       const float* __restrict__ bias,    // [L, L]
                       _Float16* __restrict__ ctx) {      // [B, L, NH*128]
  constexpr int KSTR = 136;  // 128 + 8 pad halves: frag lanes land on distinct banks
  constexpr int VSTR = 40;   // 32 + 8 pad
  constexpr int PSTR = 40;
  __shared__ __align__(16) _Float16 kbuf[2][32 * KSTR];
  __shared__ __align__(16) _Float16 vbuf[2][128 * VSTR];
  __shared__ __align__(16) _Float16 pbuf[8][16 * PSTR];

  const int lane = threadIdx.x & 31;
  const int w    = threadIdx.x >> 5;
  const int h    = blockIdx.y;
  const int b    = blockIdx.z;
  const int kvh  = h >> 2;                       // repeat_interleave: h -> h/4
  const int q0   = blockIdx.x * 128 + w * 16;

  const _Float16* Q  = qh  + ((size_t)(b * kNH  + h)   * kL + q0) * kHD;
  const _Float16* Kp = kh  +  (size_t)(b * kNKV + kvh) * kL * kHD;
  const _Float16* Vt = vth +  (size_t)(b * kNKV + kvh) * kHD * kL;

  // stage one 32-key K tile (32x128 halves) + V tile (128x32 halves):
  // 256 threads x (1+1) b128 async copies per thread per tile.
  auto stage = [&](int kb, int pg) {
    const int t = threadIdx.x;
    cp16_to_lds(Kp + (size_t)(kb + (t >> 3)) * kHD + (t & 7) * 16,
                &kbuf[pg][(t >> 3) * KSTR + (t & 7) * 16]);
    cp16_to_lds(Vt + (size_t)(t >> 1) * kL + kb + (t & 1) * 16,
                &vbuf[pg][(t >> 1) * VSTR + (t & 1) * 16]);
  };

  v16h qa[4];
#pragma unroll
  for (int t = 0; t < 4; ++t) qa[t] = load_frag_a(Q + 32 * t, kHD);

  float mrow[8], lrow[8];
  v8f o[8] = {};
#pragma unroll
  for (int j = 0; j < 8; ++j) { mrow[j] = -3.0e38f; lrow[j] = 0.0f; }

  const int g  = lane >> 4;
  const int nn = lane & 15;

  stage(0, 0);
  for (int kb = 0; kb < kL; kb += 32) {
    const int pg = (kb >> 5) & 1;
    if (kb + 32 < kL) {      // issue next tile, wait only for the previous one
      stage(kb + 32, pg ^ 1);
      wait_async_le<2>();
    } else {
      wait_async_le<0>();
    }
    __syncthreads();         // staged tile visible to all waves

    // ---- S = Q * K^T (two 16x16 tiles covering 32 keys) ----
    v8f s0 = {}, s1 = {};
#pragma unroll
    for (int t = 0; t < 4; ++t) {
      v16h k0 = load_frag_b(&kbuf[pg][32 * t], KSTR);
      v16h k1 = load_frag_b(&kbuf[pg][16 * KSTR + 32 * t], KSTR);
      s0 = WMMA_F16(qa[t], k0, s0);
      s1 = WMMA_F16(qa[t], k1, s1);
    }
    // ---- bias + online softmax (row stats via 16-lane butterflies) ----
#pragma unroll
    for (int j = 0; j < 8; ++j) {
      const int qr = q0 + g * 8 + j;
      float v0 = s0[j] + bias[(size_t)qr * kL + kb + nn];
      float v1 = s1[j] + bias[(size_t)qr * kL + kb + 16 + nn];
      float mx = fmaxf(v0, v1);
#pragma unroll
      for (int d = 1; d < 16; d <<= 1) mx = fmaxf(mx, __shfl_xor(mx, d, 32));
      const float mnew = fmaxf(mrow[j], mx);
      const float a    = __expf(mrow[j] - mnew);
      const float e0   = __expf(v0 - mnew);
      const float e1   = __expf(v1 - mnew);
      float sum = e0 + e1;
#pragma unroll
      for (int d = 1; d < 16; d <<= 1) sum += __shfl_xor(sum, d, 32);
      lrow[j] = lrow[j] * a + sum;
      mrow[j] = mnew;
#pragma unroll
      for (int t = 0; t < 8; ++t) o[t][j] *= a;    // rescale running output
      pbuf[w][(g * 8 + j) * PSTR + nn]      = (_Float16)e0;
      pbuf[w][(g * 8 + j) * PSTR + 16 + nn] = (_Float16)e1;
    }
    // per-wave LDS tile: DS ops from one wave are processed in order, so the
    // cross-lane RAW below is safe without a block barrier.
    v16h pf = load_frag_a(&pbuf[w][0], PSTR);
    // ---- O += P * V ----
#pragma unroll
    for (int t = 0; t < 8; ++t) {
      v16h vf_ = load_frag_b(&vbuf[pg][(16 * t) * VSTR], VSTR);
      o[t] = WMMA_F16(pf, vf_, o[t]);
    }
    __syncthreads();         // everyone done with buffer pg before re-staging it
  }

  float inv[8];
#pragma unroll
  for (int j = 0; j < 8; ++j) inv[j] = 1.0f / lrow[j];
#pragma unroll
  for (int t = 0; t < 8; ++t)
#pragma unroll
    for (int j = 0; j < 8; ++j) {
      const int qr = q0 + g * 8 + j;
      ctx[((size_t)b * kL + qr) * (kNH * kHD) + h * kHD + 16 * t + nn] =
          (_Float16)(o[t][j] * inv[j]);
    }
}

// ---------------------------------------------------------------------------
// Launcher
// ---------------------------------------------------------------------------
extern "C" void kernel_launch(void* const* d_in, const int* in_sizes, int n_in,
                              void* d_out, int out_size, void* d_ws, size_t ws_size,
                              hipStream_t stream) {
  (void)in_sizes; (void)n_in; (void)out_size; (void)ws_size;
  const float* x     = (const float*)d_in[0];
  const float* bias  = (const float*)d_in[1];   // [1,1,L,L]
  const float* freqs = (const float*)d_in[2];   // [L, 64]
  const float* wq    = (const float*)d_in[3];   // [2048, 2048]
  const float* wk    = (const float*)d_in[4];   // [512, 2048]
  const float* wv    = (const float*)d_in[5];   // [512, 2048]
  const float* wo    = (const float*)d_in[6];   // [2048, 2048]
  const float* qnw   = (const float*)d_in[7];
  const float* knw   = (const float*)d_in[8];
  float* out = (float*)d_out;

  const size_t M   = (size_t)kB * kL;           // 4096
  const size_t nX  = M * kC;                    // 8.39M
  const size_t nWq = (size_t)kC * kC;           // 4.19M
  const size_t nWk = (size_t)(kNKV * kHD) * kC; // 1.05M
  const size_t nQ  = M * (size_t)(kNH * kHD);
  const size_t nK  = M * (size_t)(kNKV * kHD);

  char* ws = (char*)d_ws;
  size_t off = 0;
  auto take = [&](size_t bytes) -> char* {
    char* p = ws + off;
    off = (off + bytes + 255) & ~(size_t)255;
    return p;
  };
  _Float16* xh   = (_Float16*)take(nX  * 2);
  _Float16* wqh  = (_Float16*)take(nWq * 2);
  _Float16* wkh  = (_Float16*)take(nWk * 2);
  _Float16* wvh  = (_Float16*)take(nWk * 2);
  _Float16* woh  = (_Float16*)take(nWq * 2);
  float*    qf   = (float*)   take(nQ  * 4);
  float*    kf   = (float*)   take(nK  * 4);
  float*    vf   = (float*)   take(nK  * 4);
  _Float16* qh   = (_Float16*)take(nQ  * 2);
  _Float16* kh   = (_Float16*)take(nK  * 2);
  _Float16* vth  = (_Float16*)take(nK  * 2);
  _Float16* ctxh = (_Float16*)take(nX  * 2);

  // 1) downconvert activations + weights to f16
  cvt_f32_f16_x4<<<(int)(nX  / 4 / 256), 256, 0, stream>>>(x,  xh,  (int)(nX  / 4));
  cvt_f32_f16_x4<<<(int)(nWq / 4 / 256), 256, 0, stream>>>(wq, wqh, (int)(nWq / 4));
  cvt_f32_f16_x4<<<(int)(nWk / 4 / 256), 256, 0, stream>>>(wk, wkh, (int)(nWk / 4));
  cvt_f32_f16_x4<<<(int)(nWk / 4 / 256), 256, 0, stream>>>(wv, wvh, (int)(nWk / 4));
  cvt_f32_f16_x4<<<(int)(nWq / 4 / 256), 256, 0, stream>>>(wo, woh, (int)(nWq / 4));

  // 2) QKV projections (WMMA GEMMs, f32 out)
  gemm_nt_wmma<<<dim3(kC / 128, (int)(M / 128)), 256, 0, stream>>>(xh, wqh, qf, (int)M, kC, kC);
  gemm_nt_wmma<<<dim3((kNKV * kHD) / 128, (int)(M / 128)), 256, 0, stream>>>(xh, wkh, kf, (int)M, kNKV * kHD, kC);
  gemm_nt_wmma<<<dim3((kNKV * kHD) / 128, (int)(M / 128)), 256, 0, stream>>>(xh, wvh, vf, (int)M, kNKV * kHD, kC);

  // 3) RMSNorm + RoPE -> [b,h,l,d] f16 (q folded with 1/sqrt(HD)); V -> [b,h,d,l] f16
  rmsnorm_rope_kernel<<<(int)(M * kNH  / 8), 256, 0, stream>>>(qf, qnw, freqs, qh, kNH,  kScale);
  rmsnorm_rope_kernel<<<(int)(M * kNKV / 8), 256, 0, stream>>>(kf, knw, freqs, kh, kNKV, 1.0f);
  v_transpose_kernel<<<(int)(nK / 256), 256, 0, stream>>>(vf, vth);

  // 4) flash attention (async-LDS staged K/V)
  flash_attn_kernel<<<dim3(kL / 128, kNH, kB), 256, 0, stream>>>(qh, kh, vth, bias, ctxh);

  // 5) output projection -> f32 result
  gemm_nt_wmma<<<dim3(kC / 128, (int)(M / 128)), 256, 0, stream>>>(ctxh, woh, out, (int)M, kC, kC);
}
